// CxAM_11407433138297
// MI455X (gfx1250) — matrix operand order repaired and checked by reading, not compiled, not run on hardware
//
#include <hip/hip_runtime.h>
#include <cstdint>

// Problem sizes (fixed by reference)
#define B_  16
#define C_  256
#define HW_ 4096
#define CR_ 32
#define NT_ 64   // spatial (N) tile per workgroup

typedef __attribute__((ext_vector_type(16))) _Float16     v16h;
typedef __attribute__((ext_vector_type(8)))  _Float16     v8h;
typedef __attribute__((ext_vector_type(8)))  float        v8f;
typedef __attribute__((ext_vector_type(4)))  unsigned int v4u;
typedef __attribute__((ext_vector_type(8)))  int          v8i;
typedef __attribute__((ext_vector_type(4)))  int          v4i;

union H16 { v16h v; v8h h[2]; };

// ---------------------------------------------------------------------------
// Kernel 1: x_mean[b,c] = mean over HW of x[b,c,:,:]   (one block per row)
// ---------------------------------------------------------------------------
__global__ __launch_bounds__(256)
void cxam_rowmean(const float* __restrict__ x, float* __restrict__ xm) {
#if defined(__HIP_DEVICE_COMPILE__)
    __shared__ float red[256];
    const int blk = blockIdx.x;                 // b*C + c
    const float* row = x + (size_t)blk * HW_;
    float s = 0.f;
    for (int i = threadIdx.x; i < HW_; i += 256) s += row[i];
    red[threadIdx.x] = s;
    __syncthreads();
    for (int st = 128; st > 0; st >>= 1) {
        if ((int)threadIdx.x < st) red[threadIdx.x] += red[threadIdx.x + st];
        __syncthreads();
    }
    if (threadIdx.x == 0) xm[blk] = red[0] * (1.0f / (float)HW_);
#endif
}

// ---------------------------------------------------------------------------
// Kernel 2 (tiny, 1 block): k_mean = Wk@x_mean+bk ; w_att = Wq^T k_mean ;
// bias_att = bq . k_mean ; Wvh = (f16)Wv
// ---------------------------------------------------------------------------
__global__ __launch_bounds__(256)
void cxam_prep(const float* __restrict__ xm, const float* __restrict__ Wq,
               const float* __restrict__ bq, const float* __restrict__ Wk,
               const float* __restrict__ bk, const float* __restrict__ Wv,
               float* __restrict__ km, float* __restrict__ watt,
               float* __restrict__ batt, _Float16* __restrict__ Wvh) {
#if defined(__HIP_DEVICE_COMPILE__)
    const int t = threadIdx.x;
    for (int idx = t; idx < B_ * CR_; idx += 256) {
        int b = idx / CR_, o = idx % CR_;
        float s = bk[o];
        for (int c = 0; c < C_; ++c) s += Wk[o * C_ + c] * xm[b * C_ + c];
        km[idx] = s;
    }
    __syncthreads();
    for (int idx = t; idx < B_ * C_; idx += 256) {
        int b = idx / C_, c = idx % C_;
        float s = 0.f;
        for (int o = 0; o < CR_; ++o) s += km[b * CR_ + o] * Wq[o * C_ + c];
        watt[idx] = s;
    }
    if (t < B_) {
        float s = 0.f;
        for (int o = 0; o < CR_; ++o) s += bq[o] * km[t * CR_ + o];
        batt[t] = s;
    }
    for (int i = t; i < C_ * C_; i += 256) Wvh[i] = (_Float16)Wv[i];
#endif
}

// ---------------------------------------------------------------------------
// Kernel 3: fused  out[b,o,p] = (Wv@x + bv) * sigmoid(w_att.x[:,p] + bias)
// WG = 256 threads (8 waves). WG tile: 256(M=o) x 64(N=p), K=256 (=C).
// Each wave: TDM-stages a 32x64 fp32 slab of x into LDS; threads fuse the
// attention dot from the staged tile; convert+swizzle to fp16 B-fragment
// layout; wave w computes M rows [32w,32w+32) with v_wmma_f32_16x16x32_f16.
// __launch_bounds__(256,4) + K-loop unroll capped at 2: keeps steady-state
// live set ~180 VGPRs -> no spills, no >256-VGPR MSB path, 2 WGs per WGP.
// ---------------------------------------------------------------------------
__global__ __launch_bounds__(256, 4)
void cxam_gemm(const float* __restrict__ x, const _Float16* __restrict__ Wvh,
               const float* __restrict__ bv, const float* __restrict__ watt,
               const float* __restrict__ batt, float* __restrict__ out) {
#if defined(__HIP_DEVICE_COMPILE__)
    __shared__ float    stage[C_ * NT_];        // 64 KB : fp32 x tile [k][n]
    __shared__ _Float16 sw[C_ * NT_];           // 32 KB : fp16, B-fragment order
    __shared__ float    attred[4][NT_];
    __shared__ float    attv[NT_];
    __shared__ float    bvs[C_];

    const int t    = threadIdx.x;
    const int lane = t & 31;
    const int hi   = lane >> 4;                 // K-half select (ISA A/B layout)
    const int ln   = lane & 15;
    const int b    = blockIdx.y;
    const int p0   = blockIdx.x * NT_;
    const int wv   = __builtin_amdgcn_readfirstlane(t >> 5);  // wave id 0..7

    bvs[t] = bv[t];

    // ---- TDM: wave wv DMAs x[b, 32wv:32wv+32, p0:p0+64] (fp32) into LDS ----
    {
        const float* gsrc = x + ((size_t)b * C_ + (size_t)(wv * 32)) * HW_ + p0;
        unsigned long long ga = (unsigned long long)(uintptr_t)gsrc;
        unsigned lds_off = (unsigned)(size_t)(void*)stage + (unsigned)(wv * 32 * NT_ * 4);

        v4u g0;
        g0.x = 1u;                                     // count=1 (valid user D#)
        g0.y = lds_off;                                // lds_addr (bytes)
        g0.z = (unsigned)(ga & 0xFFFFFFFFu);           // global_addr[31:0]
        g0.w = (unsigned)((ga >> 32) & 0x01FFFFFFu) | (2u << 30); // addr[56:32]|type=2

        v8i g1;
        g1[0] = (int)(2u << 16);                       // data_size = 4 bytes
        g1[1] = (int)(((unsigned)HW_ & 0xFFFFu) << 16);// tensor_dim0[15:0]
        g1[2] = (int)((HW_ >> 16) | (32u << 16));      // td0[31:16] | tensor_dim1[15:0]=32
        g1[3] = (int)(NT_ << 16);                      // td1[31:16]=0 | tile_dim0=64
        g1[4] = 32;                                    // tile_dim1=32 | tile_dim2=0
        g1[5] = HW_;                                   // tensor_dim0_stride[31:0]
        g1[6] = (int)(((unsigned)HW_ & 0xFFFFu) << 16);// td0s[47:32]=0 | td1_stride[15:0]
        g1[7] = (int)(HW_ >> 16);                      // td1_stride[47:16]

        v4i zz4 = {0, 0, 0, 0};
        v8i zz8 = {0, 0, 0, 0, 0, 0, 0, 0};
        __builtin_amdgcn_tensor_load_to_lds(g0, g1, zz4, zz4, zz8, 0);
        __builtin_amdgcn_s_wait_tensorcnt(0);
    }
    __syncthreads();

    // ---- fused attention partial: 4 threads per position, 64 K each ----
    {
        const int pp = t & (NT_ - 1);
        const int part = t >> 6;                       // 0..3
        float s = 0.f;
        const float* wa = watt + b * C_ + part * 64;
        const float* sp = stage + part * 64 * NT_ + pp;
        #pragma unroll 8
        for (int k = 0; k < 64; ++k) s += wa[k] * sp[k * NT_];
        attred[part][pp] = s;
    }

    // ---- fp32 -> fp16 swizzle into WMMA B-fragment order ----
    // B 32x16 f16 fragment: lanes 0-15 hold N=lane, K=0..15; lanes 16-31 hold
    // N=lane-16, K=16..31; each lane's 16 halves packed contiguously.
    for (int i = 0; i < (C_ * NT_) / 256; ++i) {
        int lin = t + 256 * i;
        int k = lin >> 6;                              // 0..255
        int n = lin & (NT_ - 1);                       // 0..63
        float vx = stage[lin];
        int slab = k >> 5, kk = k & 31;                // K-slab of 32
        int nt = n >> 4, nn = n & 15;                  // N-tile of 16
        int l2 = nn + ((kk >> 4) << 4);                // dest lane
        int hf = kk & 15;                              // dest half
        sw[(((slab * 4 + nt) * 32 + l2) << 4) + hf] = (_Float16)vx;
    }
    __syncthreads();

    if (t < NT_) {
        float s = batt[b] + attred[0][t] + attred[1][t] + attred[2][t] + attred[3][t];
        attv[t] = 1.0f / (1.0f + expf(-s));
    }
    __syncthreads();

    // ---- WMMA main loop: wave wv -> M rows [32wv, 32wv+32), all 64 N ----
    const int m0 = wv * 32;
    v8f acc[2][4] = {};
    #pragma unroll 2
    for (int s = 0; s < C_ / 32; ++s) {
        H16 a[2];
        #pragma unroll
        for (int Mt = 0; Mt < 2; ++Mt) {
            // A 16x32 f16 layout: lanes0-15 row=ln K=s*32+[0..15],
            // lanes16-31 row=ln K=s*32+[16..31]  (row-major coalesced)
            const _Float16* ap = Wvh + (size_t)(m0 + 16 * Mt + ln) * C_ + s * 32 + hi * 16;
            a[Mt].h[0] = *(const v8h*)ap;
            a[Mt].h[1] = *(const v8h*)(ap + 8);
        }
        #pragma unroll
        for (int f = 0; f < 4; ++f) {
            H16 bb;
            const _Float16* bp = &sw[(((s * 4 + f) * 32 + lane) << 4)];
            bb.h[0] = *(const v8h*)bp;                 // ds_read_b128
            bb.h[1] = *(const v8h*)(bp + 8);           // ds_read_b128
            #pragma unroll
            for (int Mt = 0; Mt < 2; ++Mt)
                acc[Mt][f] = __builtin_amdgcn_wmma_f32_16x16x32_f16(
                    false, a[Mt].v, false, bb.v, (short)0, acc[Mt][f], false, false);
        }
    }

    // ---- epilogue: C/D 16x16 f32 layout: VGPR r -> M=r+8*hi, N=ln ----
    #pragma unroll
    for (int Mt = 0; Mt < 2; ++Mt) {
        #pragma unroll
        for (int f = 0; f < 4; ++f) {
            const int ob = m0 + 16 * Mt + 8 * hi;
            const int pl = 16 * f + ln;
            const float av = attv[pl];
            float* op = out + ((size_t)b * C_ + ob) * HW_ + p0 + pl;
            #pragma unroll
            for (int r = 0; r < 8; ++r)
                op[(size_t)r * HW_] = (acc[Mt][f][r] + bvs[ob + r]) * av;
        }
    }
#endif
}

// ---------------------------------------------------------------------------
extern "C" void kernel_launch(void* const* d_in, const int* in_sizes, int n_in,
                              void* d_out, int out_size, void* d_ws, size_t ws_size,
                              hipStream_t stream) {
    const float* x  = (const float*)d_in[0];
    const float* Wq = (const float*)d_in[1];
    const float* bq = (const float*)d_in[2];
    const float* Wk = (const float*)d_in[3];
    const float* bk = (const float*)d_in[4];
    const float* Wv = (const float*)d_in[5];
    const float* bv = (const float*)d_in[6];
    float* out = (float*)d_out;

    char* ws = (char*)d_ws;
    float*    xm   = (float*)(ws);               // 16 KB  x_mean[B,C]
    float*    km   = (float*)(ws + 16 * 1024);   // 2 KB   k_mean[B,CR]
    float*    watt = (float*)(ws + 20 * 1024);   // 16 KB  w_att[B,C]
    float*    batt = (float*)(ws + 36 * 1024);   // 64 B   bias_att[B]
    _Float16* Wvh  = (_Float16*)(ws + 40 * 1024);// 128 KB f16 Wv

    cxam_rowmean<<<dim3(B_ * C_), dim3(256), 0, stream>>>(x, xm);
    cxam_prep<<<dim3(1), dim3(256), 0, stream>>>(xm, Wq, bq, Wk, bk, Wv,
                                                 km, watt, batt, Wvh);
    cxam_gemm<<<dim3(HW_ / NT_, B_), dim3(256), 0, stream>>>(x, Wvh, bv, watt,
                                                             batt, out);
}
